// AsymAttentionLayer_85727547228247
// MI455X (gfx1250) — matrix operand order repaired and checked by reading, not compiled, not run on hardware
//
#include <hip/hip_runtime.h>

#define DIM   512
#define SEQ   32
#define NTOK  65536      // B*ch*num_p tokens per stage
#define NSEQ  2048       // sequences per stage (NTOK/SEQ)

typedef __attribute__((ext_vector_type(16))) __bf16 v16bf;
typedef __attribute__((ext_vector_type(8)))  float  v8f;

struct U4 { unsigned x, y, z, w; };          // POD 16-byte chunk
union FragAB { v16bf v; U4 q[2]; unsigned short us[16]; };

__device__ __forceinline__ unsigned short f2bf(float f) {
  union { float f; unsigned u; } a; a.f = f;
  unsigned u = a.u;
  unsigned r = u + 0x7FFFu + ((u >> 16) & 1u);   // round-to-nearest-even
  return (unsigned short)(r >> 16);
}

// ---------------------------------------------------------------------------
// fp32 -> bf16 weight conversion (W is (N,K) row-major: already B-fragment order)
// ---------------------------------------------------------------------------
__global__ void k_convert(const float* __restrict__ src,
                          unsigned short* __restrict__ dst, int n) {
  int i = (blockIdx.x * 256 + threadIdx.x) * 8;
  if (i + 7 >= n + 7) return;
  union { unsigned short h[8]; U4 q; } o;
#pragma unroll
  for (int j = 0; j < 8; ++j) o.h[j] = f2bf(src[i + j]);
  *(U4*)(dst + i) = o.q;
}

// ---------------------------------------------------------------------------
// x [B,ch,np,dim] fp32  ->  X1 [NSEQ*SEQ, DIM] bf16 with stage-1 permutation:
//   token T = ((b*32+ng)*8+np)*32 + gi  <-  x[b, ng*32+gi, np, :]
// ---------------------------------------------------------------------------
__global__ void k_permute_x(const float* __restrict__ x,
                            unsigned short* __restrict__ X1) {
  int tid = blockIdx.x * 256 + threadIdx.x;          // NTOK*64 threads
  int c  = (tid & 63) * 8;
  int T  = tid >> 6;
  int gi = T & 31, np = (T >> 5) & 7, ng = (T >> 8) & 31, b = T >> 13;
  const float* src =
      x + ((((size_t)b * 1024 + (size_t)ng * 32 + gi) * 8 + np) * 512 + c);
  union { unsigned short h[8]; U4 q; } o;
#pragma unroll
  for (int j = 0; j < 8; ++j) o.h[j] = f2bf(src[j]);
  *(U4*)(X1 + (size_t)T * DIM + c) = o.q;
}

// ---------------------------------------------------------------------------
// C[M,512] = A[M,512] * W[512,512]^T + bias   (bf16 in, f32 acc, bf16/f32 out)
// block = 256 thr (8 waves), block tile 128x128, wave tile 32x64
// ---------------------------------------------------------------------------
template <bool F32OUT>
__global__ __launch_bounds__(256)
void k_gemm(const unsigned short* __restrict__ A,
            const unsigned short* __restrict__ W,
            const float* __restrict__ bias, void* __restrict__ outp) {
  const int lane = threadIdx.x & 31;
  const int wave = threadIdx.x >> 5;
  const int m16  = lane & 15, half = lane >> 4;
  const int RM   = blockIdx.x * 128 + (wave & 3) * 32;
  const int CN   = blockIdx.y * 128 + (wave >> 2) * 64;

  const v8f z = {0.f, 0.f, 0.f, 0.f, 0.f, 0.f, 0.f, 0.f};
  v8f acc[2][4];
#pragma unroll
  for (int i = 0; i < 2; ++i)
#pragma unroll
    for (int j = 0; j < 4; ++j) acc[i][j] = z;

  const unsigned short* aptr[2];
#pragma unroll
  for (int mt = 0; mt < 2; ++mt)
    aptr[mt] = A + (size_t)(RM + mt * 16 + m16) * DIM + half * 8;
  const unsigned short* bptr[4];
#pragma unroll
  for (int nt = 0; nt < 4; ++nt)
    bptr[nt] = W + (size_t)(CN + nt * 16 + m16) * DIM + half * 8;

  for (int kk = 0; kk < DIM; kk += 32) {
    FragAB a[2], b[4];
#pragma unroll
    for (int mt = 0; mt < 2; ++mt) {
      a[mt].q[0] = *(const U4*)(aptr[mt] + kk);        // K = half*8 .. +7
      a[mt].q[1] = *(const U4*)(aptr[mt] + kk + 16);   // K = 16+half*8 .. +7
      __builtin_prefetch(aptr[mt] + kk + 32, 0, 1);    // global_prefetch_b8
    }
#pragma unroll
    for (int nt = 0; nt < 4; ++nt) {
      b[nt].q[0] = *(const U4*)(bptr[nt] + kk);        // K = half*8 .. +7
      b[nt].q[1] = *(const U4*)(bptr[nt] + kk + 16);   // K = 16+half*8 .. +7
    }
#pragma unroll
    for (int mt = 0; mt < 2; ++mt)
#pragma unroll
      for (int nt = 0; nt < 4; ++nt)
        acc[mt][nt] = __builtin_amdgcn_wmma_f32_16x16x32_bf16(
            false, a[mt].v, false, b[nt].v, (short)0, acc[mt][nt], false, false);
  }

#pragma unroll
  for (int nt = 0; nt < 4; ++nt) {
    const int col = CN + nt * 16 + m16;
    const float bv = bias[col];
#pragma unroll
    for (int mt = 0; mt < 2; ++mt)
#pragma unroll
      for (int r = 0; r < 8; ++r) {
        const int row = RM + mt * 16 + r + half * 8;   // C layout: M = r + 8*half
        const float val = acc[mt][nt][r] + bv;
        if (F32OUT)
          ((float*)outp)[(size_t)row * DIM + col] = val;
        else
          ((unsigned short*)outp)[(size_t)row * DIM + col] = f2bf(val);
      }
  }
}

// ---------------------------------------------------------------------------
// Attention over one sequence of 32 tokens per block; one wave per head.
// S = QK^T (WMMA), softmax via LDS, O = P*V (WMMA), output store fuses the
// inter-stage permutation.
// ---------------------------------------------------------------------------
#define PSTR 48
template <int STAGE>
__global__ __launch_bounds__(256)
void k_attn(const unsigned short* __restrict__ Q,
            const unsigned short* __restrict__ K,
            const unsigned short* __restrict__ V,
            unsigned short* __restrict__ O) {
  __shared__ float          S[8][32][33];
  __shared__ unsigned short P[8][32][PSTR];

  const int lane = threadIdx.x & 31;
  const int h    = threadIdx.x >> 5;          // head = wave id
  const int m16  = lane & 15, half = lane >> 4;
  const int r    = blockIdx.x;                // sequence id
  const size_t seqbase = (size_t)r * SEQ;

  const v8f z = {0.f, 0.f, 0.f, 0.f, 0.f, 0.f, 0.f, 0.f};

  // ---- S = Q * K^T  (E = 64 -> two K-chunks of 32) ----
  v8f sacc[2][2] = {{z, z}, {z, z}};
#pragma unroll
  for (int ec = 0; ec < 2; ++ec) {
    FragAB qa[2], kb[2];
#pragma unroll
    for (int mt = 0; mt < 2; ++mt) {
      const unsigned short* p =
          Q + (seqbase + mt * 16 + m16) * DIM + h * 64 + ec * 32 + half * 8;
      qa[mt].q[0] = *(const U4*)p;
      qa[mt].q[1] = *(const U4*)(p + 16);
    }
#pragma unroll
    for (int nt = 0; nt < 2; ++nt) {
      const unsigned short* p =
          K + (seqbase + nt * 16 + m16) * DIM + h * 64 + ec * 32 + half * 8;
      kb[nt].q[0] = *(const U4*)p;
      kb[nt].q[1] = *(const U4*)(p + 16);
    }
#pragma unroll
    for (int mt = 0; mt < 2; ++mt)
#pragma unroll
      for (int nt = 0; nt < 2; ++nt)
        sacc[mt][nt] = __builtin_amdgcn_wmma_f32_16x16x32_bf16(
            false, qa[mt].v, false, kb[nt].v, (short)0, sacc[mt][nt], false, false);
  }

  // ---- dump S to LDS ----
#pragma unroll
  for (int mt = 0; mt < 2; ++mt)
#pragma unroll
    for (int nt = 0; nt < 2; ++nt)
#pragma unroll
      for (int rr = 0; rr < 8; ++rr)
        S[h][mt * 16 + rr + half * 8][nt * 16 + m16] = sacc[mt][nt][rr];
  __syncthreads();

  // ---- softmax(S / sqrt(64)): lane owns row=lane ----
  {
    float* row = S[h][lane];
    float mx = -1e30f;
#pragma unroll
    for (int j = 0; j < 32; ++j) {
      float s = row[j] * 0.125f;
      row[j] = s;
      mx = s > mx ? s : mx;
    }
    float sum = 0.f;
#pragma unroll
    for (int j = 0; j < 32; ++j) {
      float e = __expf(row[j] - mx);
      row[j] = e;
      sum += e;
    }
    const float inv = 1.f / sum;
#pragma unroll
    for (int j = 0; j < 32; ++j) P[h][lane][j] = f2bf(row[j] * inv);
  }
  __syncthreads();

  // ---- O = P * V  (K = 32 tokens, one chunk) ----
  FragAB pa[2], vb[4];
#pragma unroll
  for (int mt = 0; mt < 2; ++mt) {
    const unsigned short* p = &P[h][mt * 16 + m16][half * 8];
    pa[mt].q[0] = *(const U4*)p;            // 16B aligned (PSTR*2 = 96)
    pa[mt].q[1] = *(const U4*)(p + 16);
  }
#pragma unroll
  for (int nt = 0; nt < 4; ++nt) {
#pragma unroll
    for (int e = 0; e < 16; ++e)            // B-frag: K=(e&7)+16*(e>>3)+8*half, N=m16 (V^T gather)
      vb[nt].us[e] =
          V[(seqbase + (e & 7) + (e >> 3) * 16 + half * 8) * DIM + h * 64 + nt * 16 + m16];
  }
  v8f oacc[2][4];
#pragma unroll
  for (int mt = 0; mt < 2; ++mt)
#pragma unroll
    for (int nt = 0; nt < 4; ++nt)
      oacc[mt][nt] = __builtin_amdgcn_wmma_f32_16x16x32_bf16(
          false, pa[mt].v, false, vb[nt].v, (short)0, z, false, false);

  // ---- store with fused inter-stage permutation ----
  const int np = r & 7, sg = (r >> 3) & 31, b = r >> 8;
#pragma unroll
  for (int mt = 0; mt < 2; ++mt)
#pragma unroll
    for (int nt = 0; nt < 4; ++nt) {
      const int col = h * 64 + nt * 16 + m16;
#pragma unroll
      for (int rr = 0; rr < 8; ++rr) {
        const int t = mt * 16 + rr + half * 8;   // token within sequence
        size_t dst;
        if (STAGE == 1) {
          // seq r = (b, ng=sg, np), token t = gi  ->  X2 row (b,gi,np), tok ng
          dst = (((((size_t)b * 32 + t) * 8 + np) * 32 + sg) * DIM) + col;
        } else {
          // seq r = (b, gi=sg, np), token t = ng  ->  Y[b, ng*32+gi, np, :]
          dst = ((((size_t)b * 1024 + (size_t)t * 32 + sg) * 8 + np) * DIM) + col;
        }
        O[dst] = f2bf(oacc[mt][nt][rr]);
      }
    }
}

// ---------------------------------------------------------------------------
extern "C" void kernel_launch(void* const* d_in, const int* in_sizes, int n_in,
                              void* d_out, int out_size, void* d_ws, size_t ws_size,
                              hipStream_t stream) {
  (void)in_sizes; (void)n_in; (void)out_size; (void)ws_size;
  const float* x   = (const float*)d_in[0];
  const float* w[7]  = {(const float*)d_in[1], (const float*)d_in[3],
                        (const float*)d_in[5], (const float*)d_in[7],
                        (const float*)d_in[9], (const float*)d_in[11],
                        (const float*)d_in[13]};
  const float* bias[7] = {(const float*)d_in[2], (const float*)d_in[4],
                          (const float*)d_in[6], (const float*)d_in[8],
                          (const float*)d_in[10], (const float*)d_in[12],
                          (const float*)d_in[14]};

  unsigned short* wb[7];
  for (int i = 0; i < 7; ++i)
    wb[i] = (unsigned short*)d_ws + (size_t)i * (DIM * DIM);

  unsigned short* act  = (unsigned short*)((char*)d_ws + (4ull << 20));
  const size_t    BUF  = (size_t)NTOK * DIM;          // 33.5M bf16 = 64 MB
  unsigned short* buf0 = act;
  unsigned short* buf1 = act + BUF;
  unsigned short* buf2 = act + 2 * BUF;
  unsigned short* buf3 = act + 3 * BUF;

  for (int i = 0; i < 7; ++i)
    k_convert<<<128, 256, 0, stream>>>(w[i], wb[i], DIM * DIM);

  k_permute_x<<<NTOK * 64 / 256, 256, 0, stream>>>(x, buf0);

  dim3 gg(NTOK / 128, DIM / 128);
  // stage 1: QKV projections + attention (writes permuted X2 into buf0)
  k_gemm<false><<<gg, 256, 0, stream>>>(buf0, wb[0], bias[0], buf1);
  k_gemm<false><<<gg, 256, 0, stream>>>(buf0, wb[1], bias[1], buf2);
  k_gemm<false><<<gg, 256, 0, stream>>>(buf0, wb[2], bias[2], buf3);
  k_attn<1><<<NSEQ, 256, 0, stream>>>(buf1, buf2, buf3, buf0);
  // stage 2: QKV projections + attention (writes permuted Y into buf0)
  k_gemm<false><<<gg, 256, 0, stream>>>(buf0, wb[3], bias[3], buf1);
  k_gemm<false><<<gg, 256, 0, stream>>>(buf0, wb[4], bias[4], buf2);
  k_gemm<false><<<gg, 256, 0, stream>>>(buf0, wb[5], bias[5], buf3);
  k_attn<2><<<NSEQ, 256, 0, stream>>>(buf1, buf2, buf3, buf0);
  // output projection, fp32 into d_out
  k_gemm<true><<<gg, 256, 0, stream>>>(buf0, wb[6], bias[6], d_out);
}
